// NodeCriticalityGNN_4595615006784
// MI455X (gfx1250) — compile-verified, matrix operand order checked by reading, not run on hardware
//
#include <hip/hip_runtime.h>
#include <hip/hip_bf16.h>
#include <math.h>

// Problem constants (from reference)
#define NN   100000
#define EE   1200000
#define CC   64
#define HH   4
#define DD   16
#define LL   3
#define RR   2
#define FINN 27
#define EDD  8
#define NTILES (NN / 16)     // 6250, exact
#define TPW   4              // row-tiles per wave in the WMMA GEMM

typedef __attribute__((ext_vector_type(16))) _Float16 v16h;
typedef __attribute__((ext_vector_type(8)))  float    v8f;

__device__ __forceinline__ float gelu_exact(float x) {
  // jax.nn.gelu(approximate=False): 0.5*x*(1+erf(x/sqrt(2)))
  return 0.5f * x * (1.0f + erff(x * 0.70710678118654752f));
}
// order-preserving float<->uint mapping for atomicMax on floats
__device__ __forceinline__ unsigned f32_order(float f) {
  unsigned u = __float_as_uint(f);
  return (u & 0x80000000u) ? ~u : (u | 0x80000000u);
}
__device__ __forceinline__ float order_f32(unsigned u) {
  return (u & 0x80000000u) ? __uint_as_float(u & 0x7FFFFFFFu)
                           : __uint_as_float(~u);
}

__global__ void fill_u32_kernel(unsigned* __restrict__ p, unsigned v, int n) {
  int i = blockIdx.x * blockDim.x + threadIdx.x;
  if (i < n) p[i] = v;
}

// h = gelu(layernorm(x @ w_in + b_in)).  One wave per node; lane owns cols {lane, lane+32}.
__global__ __launch_bounds__(256) void input_proj_kernel(
    const float* __restrict__ x, const float* __restrict__ w_in,
    const float* __restrict__ b_in, const float* __restrict__ lnw,
    const float* __restrict__ lnb, float* __restrict__ h) {
  __shared__ float Wl[FINN * CC];
  int tid = threadIdx.x;
  for (int i = tid; i < FINN * CC; i += 256) Wl[i] = w_in[i];
  __syncthreads();
  int wave = tid >> 5, lane = tid & 31;
  int n = blockIdx.x * 8 + wave;
  if (n >= NN) return;
  float xv = (lane < FINN) ? x[n * FINN + lane] : 0.0f;
  float a0 = b_in[lane], a1 = b_in[lane + 32];
  #pragma unroll
  for (int j = 0; j < FINN; ++j) {
    float xj = __shfl(xv, j, 32);
    a0 = fmaf(xj, Wl[j * CC + lane], a0);
    a1 = fmaf(xj, Wl[j * CC + lane + 32], a1);
  }
  float s = a0 + a1, ss = a0 * a0 + a1 * a1;
  #pragma unroll
  for (int m = 16; m; m >>= 1) { s += __shfl_xor(s, m, 32); ss += __shfl_xor(ss, m, 32); }
  float mu  = s * (1.0f / 64.0f);
  float var = ss * (1.0f / 64.0f) - mu * mu;
  float inv = rsqrtf(var + 1e-5f);
  h[n * CC + lane]      = gelu_exact((a0 - mu) * inv * lnw[lane]      + lnb[lane]);
  h[n * CC + lane + 32] = gelu_exact((a1 - mu) * inv * lnw[lane + 32] + lnb[lane + 32]);
}

// Pre-pack all 6 conv_w matrices into per-lane B-fragment order (f16):
// wpack[lr][f][lane][j], f = kbIdx*4 + nt, 16 contiguous halves per lane (32 B aligned).
// Fragment element j of lane (lh,hi): K = kbIdx*32 + hi*8 + (j&7) + (j>=8 ? 16 : 0),
// N = nt*16 + lh  -> matches the 16-bit B operand layout of V_WMMA_F32_16X16X32_F16.
__global__ void wpack_kernel(const float* __restrict__ conv_w,
                             _Float16* __restrict__ wpack) {
  int lr = blockIdx.x;                       // 0..5
  int t  = threadIdx.x;                      // 256 = 8 fragments x 32 lanes
  int f = t >> 5, lane = t & 31;
  int lh = lane & 15, hi = (lane >> 4) & 1;
  int kbIdx = f >> 2, nt = f & 3;
  const float* W = conv_w + (size_t)lr * CC * CC;
  _Float16* o = wpack + (((size_t)lr * 8 + f) * 32 + lane) * 16;
  #pragma unroll
  for (int j = 0; j < 16; ++j) {
    int kk = kbIdx * 32 + hi * 8 + (j & 7) + ((j >> 3) * 16);
    o[j] = (_Float16)W[kk * CC + nt * 16 + lh];
  }
}

// hp = h @ W (100000x64 @ 64x64), f16 WMMA / f32 accumulate.
// B fragments held in registers (loaded once per wave, 2x global_load_b128 each,
// L2-resident); each wave sweeps TPW=4 consecutive 16-row tiles -> 32 WMMAs/wave.
__global__ __launch_bounds__(256) void hp_wmma_kernel(
    const float* __restrict__ h, const _Float16* __restrict__ wpack,
    float* __restrict__ hp) {
  int lane = threadIdx.x & 31;
  int w = (blockIdx.x * 256 + (int)threadIdx.x) >> 5;   // global wave id
  int lh = lane & 15, hi = (lane >> 4) & 1;
  v16h bf[8];
  #pragma unroll
  for (int f = 0; f < 8; ++f)
    bf[f] = *(const v16h*)(wpack + ((size_t)f * 32 + lane) * 16);
  int tile0 = w * TPW;
  #pragma unroll
  for (int t = 0; t < TPW; ++t) {
    int tile = tile0 + t;
    if (tile >= NTILES) return;                         // wave-uniform
    const float* arow = h + (size_t)(tile * 16 + lh) * CC;
    if (tile + 1 < NTILES)                              // global_prefetch_b8
      __builtin_prefetch(h + (size_t)((tile + 1) * 16 + lh) * CC, 0, 3);
    v8f c[4] = {};
    #pragma unroll
    for (int kb = 0; kb < 2; ++kb) {
      int k0 = kb * 32 + hi * 8;
      v16h a;
      #pragma unroll
      for (int j = 0; j < 8; ++j) {
        a[j]     = (_Float16)arow[k0 + j];              // K = k0..k0+7
        a[j + 8] = (_Float16)arow[k0 + 16 + j];         // K = k0+16..k0+23
      }
      #pragma unroll
      for (int nt = 0; nt < 4; ++nt)
        c[nt] = __builtin_amdgcn_wmma_f32_16x16x32_f16(
            false, a, false, bf[kb * 4 + nt], (short)0, c[nt], false, false);
    }
    // D layout: lanes 0-15 -> M=r, lanes 16-31 -> M=8+r; N = lh
    #pragma unroll
    for (int nt = 0; nt < 4; ++nt)
      #pragma unroll
      for (int r = 0; r < 8; ++r)
        hp[(size_t)(tile * 16 + hi * 8 + r) * CC + nt * 16 + lh] = c[nt][r];
  }
}

// al_src[n,h] = sum_d hp[n,h,d]*att_src[h,d]; same for al_dst.
__global__ __launch_bounds__(256) void attn_node_kernel(
    const float* __restrict__ hp, const float* __restrict__ as_,
    const float* __restrict__ ad_, float* __restrict__ al_s,
    float* __restrict__ al_d) {
  int idx = blockIdx.x * blockDim.x + threadIdx.x;   // n*H + h
  if (idx >= NN * HH) return;
  int n = idx >> 2, hh = idx & 3;
  const float* row = hp + (size_t)n * CC + hh * DD;
  float s0 = 0.f, s1 = 0.f;
  #pragma unroll
  for (int d = 0; d < DD; ++d) {
    s0 = fmaf(row[d], as_[hh * DD + d], s0);
    s1 = fmaf(row[d], ad_[hh * DD + d], s1);
  }
  al_s[idx] = s0;
  al_d[idx] = s1;
}

// w_eff[j][h] = sum_d w_edge[j, h*16+d] * att_edge[h,d]   (folds a_e into W_e: 8x4)
__global__ void weff_kernel(const float* __restrict__ We,
                            const float* __restrict__ ae,
                            float* __restrict__ weff) {
  int t = threadIdx.x;
  if (t >= EDD * HH) return;
  int j = t >> 2, hh = t & 3;
  float s = 0.f;
  #pragma unroll
  for (int d = 0; d < DD; ++d) s = fmaf(We[j * CC + hh * DD + d], ae[hh * DD + d], s);
  weff[t] = s;   // layout [j][h]
}

// Pass 1: logits = leaky_relu(al_s[src]+al_d[dst]+eattr@w_eff), atomicMax per (dst,h).
__global__ __launch_bounds__(256) void edge_logits_kernel(
    const int* __restrict__ src, const int* __restrict__ dst,
    const float* __restrict__ ea, const float* __restrict__ al_s,
    const float* __restrict__ al_d, const float* __restrict__ weff,
    float* __restrict__ elog, unsigned* __restrict__ mmax) {
  __shared__ float wl[EDD * HH];
  if (threadIdx.x < EDD * HH) wl[threadIdx.x] = weff[threadIdx.x];
  __syncthreads();
  int e = blockIdx.x * 256 + threadIdx.x;
  if (e >= EE) return;
  int s = src[e], d = dst[e];
  float ev[EDD];
  const float* er = ea + (size_t)e * EDD;
  #pragma unroll
  for (int j = 0; j < EDD; ++j) ev[j] = er[j];
  #pragma unroll
  for (int hh = 0; hh < HH; ++hh) {
    float ale = 0.f;
    #pragma unroll
    for (int j = 0; j < EDD; ++j) ale = fmaf(ev[j], wl[j * HH + hh], ale);
    float lg = al_s[s * HH + hh] + al_d[d * HH + hh] + ale;
    lg = (lg > 0.f) ? lg : 0.2f * lg;          // leaky_relu(0.2)
    elog[e * HH + hh] = lg;
    atomicMax(&mmax[d * HH + hh], f32_order(lg));
  }
}

// Pass 2: ex = exp(logit - max[dst]); den[dst] += ex  (ex overwrites logits buffer)
__global__ __launch_bounds__(256) void edge_exp_kernel(
    const int* __restrict__ dst, const unsigned* __restrict__ mmax,
    float* __restrict__ elog, float* __restrict__ den) {
  int e = blockIdx.x * 256 + threadIdx.x;
  if (e >= EE) return;
  int d = dst[e];
  #pragma unroll
  for (int hh = 0; hh < HH; ++hh) {
    float m  = order_f32(mmax[d * HH + hh]);   // finite: dst has >=1 edge
    float ex = __expf(elog[e * HH + hh] - m);
    elog[e * HH + hh] = ex;
    atomicAdd(&den[d * HH + hh], ex);
  }
}

// Pass 3: acc[dst] += (ex/(den[dst]+1e-16)) * hp[src].  16 threads/edge, float4 each.
__global__ __launch_bounds__(256) void edge_aggr_kernel(
    const int* __restrict__ src, const int* __restrict__ dst,
    const float* __restrict__ ex, const float* __restrict__ den,
    const float* __restrict__ hp, float* __restrict__ acc) {
  int gid = blockIdx.x * 256 + threadIdx.x;
  int e = gid >> 4;
  int q = gid & 15;                             // float4 chunk: cols 4q..4q+3, head q/4
  if (e >= EE) return;
  int s = src[e], d = dst[e];
  int hh = q >> 2;
  float alpha = ex[e * HH + hh] / (den[d * HH + hh] + 1e-16f);
  float4 v = ((const float4*)(hp + (size_t)s * CC))[q];
  float* o = acc + (size_t)d * CC + q * 4;
  atomicAdd(o + 0, alpha * v.x);
  atomicAdd(o + 1, alpha * v.y);
  atomicAdd(o + 2, alpha * v.z);
  atomicAdd(o + 3, alpha * v.w);
}

// h = gelu(layernorm(0.5*(acc + b0 + b1) + h)).  Wave per node.
__global__ __launch_bounds__(256) void node_update_kernel(
    float* __restrict__ h, const float* __restrict__ acc,
    const float* __restrict__ b0, const float* __restrict__ b1,
    const float* __restrict__ lnw, const float* __restrict__ lnb) {
  int wave = threadIdx.x >> 5, lane = threadIdx.x & 31;
  int n = blockIdx.x * 8 + wave;
  if (n >= NN) return;
  int c0 = lane, c1 = lane + 32;
  float a0 = 0.5f * (acc[(size_t)n * CC + c0] + b0[c0] + b1[c0]) + h[(size_t)n * CC + c0];
  float a1 = 0.5f * (acc[(size_t)n * CC + c1] + b0[c1] + b1[c1]) + h[(size_t)n * CC + c1];
  float s = a0 + a1, ss = a0 * a0 + a1 * a1;
  #pragma unroll
  for (int m = 16; m; m >>= 1) { s += __shfl_xor(s, m, 32); ss += __shfl_xor(ss, m, 32); }
  float mu  = s * (1.0f / 64.0f);
  float var = ss * (1.0f / 64.0f) - mu * mu;
  float inv = rsqrtf(var + 1e-5f);
  h[(size_t)n * CC + c0] = gelu_exact((a0 - mu) * inv * lnw[c0] + lnb[c0]);
  h[(size_t)n * CC + c1] = gelu_exact((a1 - mu) * inv * lnw[c1] + lnb[c1]);
}

// Heads: _resmlp ends in LayerNorm over a SINGLETON axis -> exactly its bias.
// So comp = sigmoid(cnorm_b), rmav[i] = sigmoid(hnorm_b[i]) for every node (exact algebra).
__global__ void head_output_kernel(const float* __restrict__ hnorm_b,
                                   const float* __restrict__ cnorm_b,
                                   float* __restrict__ out) {
  int n = blockIdx.x * blockDim.x + threadIdx.x;
  if (n >= NN) return;
  out[n * 5 + 0] = 1.0f / (1.0f + __expf(-cnorm_b[0]));
  #pragma unroll
  for (int i = 0; i < 4; ++i)
    out[n * 5 + 1 + i] = 1.0f / (1.0f + __expf(-hnorm_b[i]));
}

extern "C" void kernel_launch(void* const* d_in, const int* in_sizes, int n_in,
                              void* d_out, int out_size, void* d_ws, size_t ws_size,
                              hipStream_t stream) {
  (void)in_sizes; (void)n_in; (void)out_size; (void)ws_size;
  const float* x        = (const float*)d_in[0];
  const int*   ei[2]    = {(const int*)d_in[1], (const int*)d_in[2]};
  const float* ea[2]    = {(const float*)d_in[3], (const float*)d_in[4]};
  const float* w_in     = (const float*)d_in[5];
  const float* b_in     = (const float*)d_in[6];
  const float* ln_in_w  = (const float*)d_in[7];
  const float* ln_in_b  = (const float*)d_in[8];
  const float* conv_w   = (const float*)d_in[9];
  const float* att_src  = (const float*)d_in[10];
  const float* att_dst  = (const float*)d_in[11];
  const float* att_edge = (const float*)d_in[12];
  const float* w_edge   = (const float*)d_in[13];
  const float* conv_b   = (const float*)d_in[14];
  const float* ln_w     = (const float*)d_in[15];
  const float* ln_b     = (const float*)d_in[16];
  const float* hnorm_b  = (const float*)d_in[22];
  const float* cnorm_b  = (const float*)d_in[30];

  // Workspace layout (~102 MB; hot set fits 192 MB L2)
  float*    h    = (float*)d_ws;
  float*    hp   = h    + (size_t)NN * CC;
  float*    acc  = hp   + (size_t)NN * CC;
  float*    al_s = acc  + (size_t)NN * CC;
  float*    al_d = al_s + (size_t)NN * HH;
  unsigned* mmax = (unsigned*)(al_d + (size_t)NN * HH);
  float*    den  = (float*)(mmax + (size_t)NN * HH);
  float*    elog = den  + (size_t)NN * HH;
  float*    weff = elog + (size_t)EE * HH;
  _Float16* wpackbuf = (_Float16*)(weff + 32);   // 6 * 4096 halves, 128B-aligned

  dim3 b256(256);
  input_proj_kernel<<<(NN + 7) / 8, b256, 0, stream>>>(x, w_in, b_in, ln_in_w, ln_in_b, h);
  wpack_kernel<<<LL * RR, b256, 0, stream>>>(conv_w, wpackbuf);

  for (int l = 0; l < LL; ++l) {
    fill_u32_kernel<<<(NN * CC + 255) / 256, b256, 0, stream>>>((unsigned*)acc, 0u, NN * CC);
    for (int r = 0; r < RR; ++r) {
      int lr = l * RR + r;
      hp_wmma_kernel<<<(NTILES + 8 * TPW - 1) / (8 * TPW), b256, 0, stream>>>(
          h, wpackbuf + (size_t)lr * 8 * 32 * 16, hp);
      attn_node_kernel<<<(NN * HH + 255) / 256, b256, 0, stream>>>(
          hp, att_src + lr * HH * DD, att_dst + lr * HH * DD, al_s, al_d);
      weff_kernel<<<1, 32, 0, stream>>>(
          w_edge + (size_t)lr * EDD * CC, att_edge + lr * HH * DD, weff);
      fill_u32_kernel<<<(NN * HH + 255) / 256, b256, 0, stream>>>(mmax, 0x007FFFFFu, NN * HH); // enc(-inf)
      fill_u32_kernel<<<(NN * HH + 255) / 256, b256, 0, stream>>>((unsigned*)den, 0u, NN * HH);
      edge_logits_kernel<<<(EE + 255) / 256, b256, 0, stream>>>(
          ei[r], ei[r] + EE, ea[r], al_s, al_d, weff, elog, mmax);
      edge_exp_kernel<<<(EE + 255) / 256, b256, 0, stream>>>(ei[r] + EE, mmax, elog, den);
      edge_aggr_kernel<<<(EE * 16) / 256, b256, 0, stream>>>(
          ei[r], ei[r] + EE, elog, den, hp, acc);
    }
    node_update_kernel<<<(NN + 7) / 8, b256, 0, stream>>>(
        h, acc, conv_b + (size_t)(l * RR + 0) * CC, conv_b + (size_t)(l * RR + 1) * CC,
        ln_w + l * CC, ln_b + l * CC);
  }

  head_output_kernel<<<(NN + 255) / 256, b256, 0, stream>>>(hnorm_b, cnorm_b, (float*)d_out);
}
